// BatchMultiHeadAttention_66168266162305
// MI455X (gfx1250) — matrix operand order repaired and checked by reading, not compiled, not run on hardware
//
#include <hip/hip_runtime.h>
#include <hip/hip_bf16.h>

typedef _Float16 v16h  __attribute__((ext_vector_type(16)));
typedef _Float16 half8 __attribute__((ext_vector_type(8)));
typedef _Float16 half4 __attribute__((ext_vector_type(4)));
typedef float    v8f   __attribute__((ext_vector_type(8)));
typedef int      v4i   __attribute__((ext_vector_type(4)));

#define BDIM 2
#define TDIM 2048
#define CDIM 1024
#define HN   16
#define HD   64
#define MTOT (BDIM*TDIM)   // 4096 tokens

// ---------------------------------------------------------------------------
// gfx1250 async global->LDS copy (ASYNCcnt-tracked), with safe fallback.
// Probe-learned signature: (int4 AS1* gsrc, int4 AS3* ldst, imm offset, imm cpol)
// ---------------------------------------------------------------------------
#if defined(__has_builtin)
#if __has_builtin(__builtin_amdgcn_global_load_async_to_lds_b128)
#define HAVE_ASYNC_LDS 1
#endif
#endif

#ifdef HAVE_ASYNC_LDS
typedef __attribute__((address_space(1))) v4i gv4i;
typedef __attribute__((address_space(3))) v4i lv4i;
#define ASYNC_CP16(dst, src)                                                   \
  __builtin_amdgcn_global_load_async_to_lds_b128((gv4i*)(src), (lv4i*)(dst),   \
                                                 0, 0)
#if __has_builtin(__builtin_amdgcn_s_wait_asynccnt)
#define ASYNC_WAIT() __builtin_amdgcn_s_wait_asynccnt(0)
#else
#define ASYNC_WAIT() asm volatile("s_wait_asynccnt 0x0" ::: "memory")
#endif
#else
#define ASYNC_CP16(dst, src) (*(half8*)(dst) = *(const half8*)(src))
#define ASYNC_WAIT() ((void)0)
#endif

// ---------------------------------------------------------------------------
// WMMA fragment loaders per CDNA5 ISA wave32 VGPR layouts.
// A (16x32 f16): lane<16 -> row=lane, K={0..7}U{16..23}; lane>=16 -> row=lane-16,
//                K={8..15}U{24..31}. base=8*kh -> two contiguous 16B runs.
// B (32x16 f16): lane -> col=lane&15; K = 8*kh + {0..7}U{16..23} -> two 16B runs.
// C/D (16x16 f32): lane -> col=lane&15; vgpr r -> row = r + 8*kh.
// ---------------------------------------------------------------------------
__device__ inline v16h ld_afrag(const _Float16* __restrict__ row, int kk, int kh) {
  const half8 lo = *(const half8*)(row + kk + 8*kh);
  const half8 hi = *(const half8*)(row + kk + 8*kh + 16);
  v16h a;
#pragma unroll
  for (int i = 0; i < 8; ++i) { a[i] = lo[i]; a[i+8] = hi[i]; }
  return a;
}

__device__ inline v16h ld_bfrag(const _Float16* __restrict__ row, int kk, int kh) {
  const half8 lo = *(const half8*)(row + kk + 8*kh);
  const half8 hi = *(const half8*)(row + kk + 8*kh + 16);
  v16h b;
#pragma unroll
  for (int i = 0; i < 8; ++i) { b[i] = lo[i]; b[i+8] = hi[i]; }
  return b;
}

__device__ inline v8f wmma16(v16h a, v16h b, v8f c) {
  return __builtin_amdgcn_wmma_f32_16x16x32_f16(false, a, false, b, (short)0, c,
                                                false, false);
}

__device__ inline v8f vzero8() {
  v8f z = {0.f,0.f,0.f,0.f,0.f,0.f,0.f,0.f};
  return z;
}

// ---------------------------------------------------------------------------
// Stage 0: f32 -> f16 conversion
// ---------------------------------------------------------------------------
__global__ void cvt_f32_f16(const float* __restrict__ src, _Float16* __restrict__ dst,
                            int n) {
  const int i = (blockIdx.x * blockDim.x + threadIdx.x) * 4;
  if (i < n) {
    const float4 v = *(const float4*)(src + i);
    half4 h;
    h[0] = (_Float16)v.x; h[1] = (_Float16)v.y;
    h[2] = (_Float16)v.z; h[3] = (_Float16)v.w;
    *(half4*)(dst + i) = h;
  }
}

// ---------------------------------------------------------------------------
// Pipelined 32(M)x64(N)-per-wave WMMA GEMM core with ping-pong fragment
// buffers: phase-1 fragments load while phase-0 WMMAs execute and vice versa.
// No register rotation copies; K unrolled by 64 (2 x 32-step phases).
// ---------------------------------------------------------------------------
__device__ inline void gemm_core(const _Float16* __restrict__ row0,
                                 const _Float16* __restrict__ row1,
                                 const _Float16* __restrict__ wrow,
                                 int kh, v8f acc[2][4]) {
  v16h a0[2], a1[2], b0[4], b1[4];
  a0[0] = ld_afrag(row0, 0, kh);
  a0[1] = ld_afrag(row1, 0, kh);
#pragma unroll
  for (int tn = 0; tn < 4; ++tn)
    b0[tn] = ld_bfrag(wrow + (size_t)tn * 16 * CDIM, 0, kh);

  for (int kk = 0; kk < CDIM; kk += 64) {
    // load phase-1 fragments (kk+32 always < CDIM)
    a1[0] = ld_afrag(row0, kk + 32, kh);
    a1[1] = ld_afrag(row1, kk + 32, kh);
#pragma unroll
    for (int tn = 0; tn < 4; ++tn)
      b1[tn] = ld_bfrag(wrow + (size_t)tn * 16 * CDIM, kk + 32, kh);

    // compute phase 0
#pragma unroll
    for (int tn = 0; tn < 4; ++tn) {
      acc[0][tn] = wmma16(a0[0], b0[tn], acc[0][tn]);
      acc[1][tn] = wmma16(a0[1], b0[tn], acc[1][tn]);
    }

    // load phase-0 fragments for next iteration (guarded harmless reload)
    const int kn = (kk + 64 < CDIM) ? (kk + 64) : 0;
    a0[0] = ld_afrag(row0, kn, kh);
    a0[1] = ld_afrag(row1, kn, kh);
#pragma unroll
    for (int tn = 0; tn < 4; ++tn)
      b0[tn] = ld_bfrag(wrow + (size_t)tn * 16 * CDIM, kn, kh);

    // compute phase 1
#pragma unroll
    for (int tn = 0; tn < 4; ++tn) {
      acc[0][tn] = wmma16(a1[0], b1[tn], acc[0][tn]);
      acc[1][tn] = wmma16(a1[1], b1[tn], acc[1][tn]);
    }
  }
}

// ---------------------------------------------------------------------------
// Stage 1: QKV projection GEMM: Out[m][n] = sum_c X[m][c] * W[n][c]
// vmode 0: store [B,H,T,HD] (Q,K).  vmode 1: store transposed [B,H,HD,T] (V).
// ---------------------------------------------------------------------------
__global__ __launch_bounds__(128)
void gemm_qkv(const _Float16* __restrict__ X, const _Float16* __restrict__ W,
              _Float16* __restrict__ Out, int vmode) {
  const int wave = threadIdx.x >> 5, lane = threadIdx.x & 31;
  const int lr = lane & 15, kh = lane >> 4;
  const int m0 = blockIdx.x * 128 + wave * 32;
  const int n0 = blockIdx.y * 64;

  v8f acc[2][4];
#pragma unroll
  for (int ms = 0; ms < 2; ++ms)
#pragma unroll
    for (int tn = 0; tn < 4; ++tn) acc[ms][tn] = vzero8();

  gemm_core(X + (size_t)(m0 + lr) * CDIM,
            X + (size_t)(m0 + 16 + lr) * CDIM,
            W + (size_t)(n0 + lr) * CDIM, kh, acc);

#pragma unroll
  for (int ms = 0; ms < 2; ++ms) {
#pragma unroll
    for (int tn = 0; tn < 4; ++tn) {
      const int n = n0 + tn * 16 + lr;
      const int h = n >> 6, d = n & (HD - 1);
#pragma unroll
      for (int r = 0; r < 8; ++r) {
        const int m = m0 + ms * 16 + r + 8 * kh;
        const int bb = m >> 11, t = m & (TDIM - 1);
        size_t idx;
        if (vmode == 0) idx = ((((size_t)bb * HN + h) * TDIM) + t) * HD + d;
        else            idx = ((((size_t)bb * HN + h) * HD) + d) * TDIM + t;
        Out[idx] = (_Float16)acc[ms][tn][r];
      }
    }
  }
}

// ---------------------------------------------------------------------------
// Stage 2: causal flash attention.
// Grid: (T/64, B*H), 4 waves/block. Per key-block j: K (64x64) and V^T (64x64)
// tiles are staged global->LDS ONCE per block via async-to-LDS, double-buffered
// so the stage of block j+1 overlaps the compute of block j. Waves own 16-query
// strips; softmax is fully register/shuffle-resident; P goes C-layout->A-layout
// through a per-wave LDS tile.
// ---------------------------------------------------------------------------
__device__ inline void stage_kv(const _Float16* __restrict__ ksrc_base,  // Kb + j*64*HD (linear 8KB)
                                const _Float16* __restrict__ vsrc_base,  // Vb + j*64   (64 rows, stride TDIM)
                                _Float16* kb, _Float16* vb, int tid) {
  // K tile: 8192B linear; 64B per thread
  const _Float16* ks = ksrc_base + tid * 32;
  _Float16*       kd = kb + tid * 32;
  ASYNC_CP16(kd,      ks);
  ASYNC_CP16(kd + 8,  ks + 8);
  ASYNC_CP16(kd + 16, ks + 16);
  ASYNC_CP16(kd + 24, ks + 24);
  // V tile: 64 rows x 128B (row stride TDIM halves); half-row per thread
  const int vr = tid >> 1, vo = (tid & 1) * 32;
  const _Float16* vs = vsrc_base + (size_t)vr * TDIM + vo;
  _Float16*       vd = vb + vr * 64 + vo;
  ASYNC_CP16(vd,      vs);
  ASYNC_CP16(vd + 8,  vs + 8);
  ASYNC_CP16(vd + 16, vs + 16);
  ASYNC_CP16(vd + 24, vs + 24);
}

__global__ __launch_bounds__(128)
void flash_attn(const _Float16* __restrict__ Q, const _Float16* __restrict__ Kg,
                const _Float16* __restrict__ Vt, _Float16* __restrict__ Aout) {
  __shared__ _Float16 kbuf[2][64 * 64];   // 2 x 8KB
  __shared__ _Float16 vbuf[2][64 * 64];   // 2 x 8KB
  __shared__ _Float16 pbuf[4][16 * 64];   // per-wave P tiles, 8KB

  const int tid = threadIdx.x;
  const int wave = tid >> 5, lane = tid & 31;
  const int lr = lane & 15, kh = lane >> 4;
  const int qblk = blockIdx.x;             // 0..T/64-1
  const int bh   = blockIdx.y;             // 0..B*H-1
  const int bb = bh / HN, hh = bh % HN;

  const _Float16* Qb = Q  + (size_t)bh * TDIM * HD;
  const _Float16* Kb = Kg + (size_t)bh * TDIM * HD;
  const _Float16* Vb = Vt + (size_t)bh * HD * TDIM;

  const int q0 = qblk * 64 + wave * 16;

  // Q A-fragments for this wave's 16 rows (d = 0..63, two K=32 chunks)
  const _Float16* qrow = Qb + (size_t)(q0 + lr) * HD;
  const v16h aq0 = ld_afrag(qrow, 0, kh);
  const v16h aq1 = ld_afrag(qrow, 32, kh);

  float mrow[8], lrow[8];
  v8f o[4];
#pragma unroll
  for (int r = 0; r < 8; ++r) { mrow[r] = -1e30f; lrow[r] = 0.f; }
#pragma unroll
  for (int tn = 0; tn < 4; ++tn) o[tn] = vzero8();

  const float scale = 0.125f;  // 1/sqrt(64)
  _Float16* pb = &pbuf[wave][0];
  const int qgb = q0 + 8 * kh;             // row of acc element r is qgb + r

  // prologue: stage key-block 0
  stage_kv(Kb, Vb, kbuf[0], vbuf[0], tid);

  for (int j = 0; j <= qblk; ++j) {
    ASYNC_WAIT();
    __syncthreads();                       // buf[cur] visible to all waves
    const int cur = j & 1;
    if (j < qblk)                          // overlap next stage with compute
      stage_kv(Kb + (size_t)(j + 1) * 64 * HD, Vb + (size_t)(j + 1) * 64,
               kbuf[1 - cur], vbuf[1 - cur], tid);

    // ---- S = Q K^T (16 x 64 per wave), K frags from LDS ----
    v8f s[4];
#pragma unroll
    for (int tn = 0; tn < 4; ++tn) {
      const _Float16* krow = &kbuf[cur][(tn * 16 + lr) * 64];
      v8f acc = vzero8();
      acc = wmma16(aq0, ld_bfrag(krow, 0, kh), acc);
      acc = wmma16(aq1, ld_bfrag(krow, 32, kh), acc);
      s[tn] = acc;
    }

    // ---- causal mask + scale (lane-local) ----
#pragma unroll
    for (int tn = 0; tn < 4; ++tn) {
      const int kg = j * 64 + tn * 16 + lr;
#pragma unroll
      for (int r = 0; r < 8; ++r)
        s[tn][r] = (kg <= qgb + r) ? s[tn][r] * scale : -1e30f;
    }

    // ---- row max: 4-tile max + xor-reduce within each 16-lane half ----
    float rmax[8];
#pragma unroll
    for (int r = 0; r < 8; ++r)
      rmax[r] = fmaxf(fmaxf(s[0][r], s[1][r]), fmaxf(s[2][r], s[3][r]));
#pragma unroll
    for (int sh = 1; sh <= 8; sh <<= 1)
#pragma unroll
      for (int r = 0; r < 8; ++r)
        rmax[r] = fmaxf(rmax[r], __shfl_xor(rmax[r], sh, 32));

    // ---- online softmax update ----
#pragma unroll
    for (int r = 0; r < 8; ++r) {
      const float mnew = fmaxf(mrow[r], rmax[r]);
      const float corr = __expf(mrow[r] - mnew);
      mrow[r] = mnew;
      lrow[r] *= corr;
#pragma unroll
      for (int tn = 0; tn < 4; ++tn) o[tn][r] *= corr;
    }

    // ---- P = exp(S - m): row sums + f16 store to per-wave LDS tile ----
    float rsum[8];
#pragma unroll
    for (int r = 0; r < 8; ++r) rsum[r] = 0.f;
#pragma unroll
    for (int tn = 0; tn < 4; ++tn) {
#pragma unroll
      for (int r = 0; r < 8; ++r) {
        const float p = __expf(s[tn][r] - mrow[r]);
        rsum[r] += p;
        pb[(r + 8 * kh) * 64 + tn * 16 + lr] = (_Float16)p;
      }
    }
#pragma unroll
    for (int sh = 1; sh <= 8; sh <<= 1)
#pragma unroll
      for (int r = 0; r < 8; ++r)
        rsum[r] += __shfl_xor(rsum[r], sh, 32);
#pragma unroll
    for (int r = 0; r < 8; ++r) lrow[r] += rsum[r];

    // ---- O += P V: A-frags from P tile, B-frags from LDS V^T tile ----
    const _Float16* prow = pb + lr * 64;
    const v16h ap0 = ld_afrag(prow, 0, kh);
    const v16h ap1 = ld_afrag(prow, 32, kh);
#pragma unroll
    for (int tn = 0; tn < 4; ++tn) {
      const _Float16* vrow = &vbuf[cur][(tn * 16 + lr) * 64];
      o[tn] = wmma16(ap0, ld_bfrag(vrow, 0, kh), o[tn]);
      o[tn] = wmma16(ap1, ld_bfrag(vrow, 32, kh), o[tn]);
    }
  }

  // ---- epilogue: O /= l, merge heads back to [B,T,C] f16 ----
#pragma unroll
  for (int tn = 0; tn < 4; ++tn) {
#pragma unroll
    for (int r = 0; r < 8; ++r) {
      const int qg = q0 + r + 8 * kh;
      const float v = o[tn][r] / lrow[r];
      Aout[((size_t)bb * TDIM + qg) * CDIM + hh * HD + tn * 16 + lr] = (_Float16)v;
    }
  }
}

// ---------------------------------------------------------------------------
// Stage 3: output projection: Out[m][n] = sum_c A[m][c]*Wp[n][c] + bp[n], f32
// ---------------------------------------------------------------------------
__global__ __launch_bounds__(128)
void gemm_proj(const _Float16* __restrict__ A, const _Float16* __restrict__ W,
               const float* __restrict__ bias, float* __restrict__ Out) {
  const int wave = threadIdx.x >> 5, lane = threadIdx.x & 31;
  const int lr = lane & 15, kh = lane >> 4;
  const int m0 = blockIdx.x * 128 + wave * 32;
  const int n0 = blockIdx.y * 64;

  v8f acc[2][4];
#pragma unroll
  for (int ms = 0; ms < 2; ++ms)
#pragma unroll
    for (int tn = 0; tn < 4; ++tn) acc[ms][tn] = vzero8();

  gemm_core(A + (size_t)(m0 + lr) * CDIM,
            A + (size_t)(m0 + 16 + lr) * CDIM,
            W + (size_t)(n0 + lr) * CDIM, kh, acc);

#pragma unroll
  for (int ms = 0; ms < 2; ++ms) {
#pragma unroll
    for (int tn = 0; tn < 4; ++tn) {
      const int n = n0 + tn * 16 + lr;
      const float bn2 = bias[n];
#pragma unroll
      for (int r = 0; r < 8; ++r) {
        const int m = m0 + ms * 16 + r + 8 * kh;
        Out[(size_t)m * CDIM + n] = acc[ms][tn][r] + bn2;
      }
    }
  }
}

// ---------------------------------------------------------------------------
extern "C" void kernel_launch(void* const* d_in, const int* in_sizes, int n_in,
                              void* d_out, int out_size, void* d_ws, size_t ws_size,
                              hipStream_t stream) {
  const float* x  = (const float*)d_in[0];
  const float* Wq = (const float*)d_in[1];
  const float* Wk = (const float*)d_in[2];
  const float* Wv = (const float*)d_in[3];
  const float* Wp = (const float*)d_in[4];
  const float* bp = (const float*)d_in[5];

  const size_t NX = (size_t)MTOT * CDIM;           // 4,194,304
  const size_t NW = (size_t)CDIM * CDIM;           // 1,048,576
  const size_t NH = (size_t)BDIM * HN * TDIM * HD; // 4,194,304

  _Float16* p   = (_Float16*)d_ws;
  _Float16* xh  = p; p += NX;
  _Float16* wqh = p; p += NW;
  _Float16* wkh = p; p += NW;
  _Float16* wvh = p; p += NW;
  _Float16* wph = p; p += NW;
  _Float16* qh  = p; p += NH;
  _Float16* khb = p; p += NH;
  _Float16* vth = p; p += NH;
  _Float16* ah  = p; p += NH;

  // Stage 0: f16 conversion
  cvt_f32_f16<<<(int)(NX / 1024), 256, 0, stream>>>(x,  xh,  (int)NX);
  cvt_f32_f16<<<(int)(NW / 1024), 256, 0, stream>>>(Wq, wqh, (int)NW);
  cvt_f32_f16<<<(int)(NW / 1024), 256, 0, stream>>>(Wk, wkh, (int)NW);
  cvt_f32_f16<<<(int)(NW / 1024), 256, 0, stream>>>(Wv, wvh, (int)NW);
  cvt_f32_f16<<<(int)(NW / 1024), 256, 0, stream>>>(Wp, wph, (int)NW);

  // Stage 1: Q/K/V projections (V stored transposed per head)
  dim3 gG(MTOT / 128, CDIM / 64);
  gemm_qkv<<<gG, 128, 0, stream>>>(xh, wqh, qh,  0);
  gemm_qkv<<<gG, 128, 0, stream>>>(xh, wkh, khb, 0);
  gemm_qkv<<<gG, 128, 0, stream>>>(xh, wvh, vth, 1);

  // Stage 2: causal flash attention (async-staged K/V, double-buffered LDS)
  flash_attn<<<dim3(TDIM / 64, BDIM * HN), 128, 0, stream>>>(qh, khb, vth, ah);

  // Stage 3: output projection + bias (f32 out)
  gemm_proj<<<gG, 128, 0, stream>>>(ah, wph, bp, (float*)d_out);
}
